// FeatureGCNProcessor_50989851738543
// MI455X (gfx1250) — matrix operand order repaired and checked by reading, compiled, not run on hardware
//
#include <hip/hip_runtime.h>

#define B_   2
#define C_   256
#define N_   9216      // 96*96
#define NTOT (B_ * N_)
#define F1_  512
#define F2_  256
#define TPB  576       // 16-row/col tiles per batch (9216/16), even

typedef __attribute__((ext_vector_type(16))) __bf16 v16bf;
typedef __attribute__((ext_vector_type(8)))  __bf16 v8bf;
typedef __attribute__((ext_vector_type(8)))  float  v8f;

__device__ __forceinline__ unsigned short f2bf(float f) {
  unsigned u = __float_as_uint(f);
  u = (u + 0x7FFFu + ((u >> 16) & 1u)) >> 16;   // round-to-nearest-even
  return (unsigned short)u;
}

// 16-bit A fragment 16x32: lane r<16 -> K k0..k0+15, lane r+16 -> k0+16..k0+31.
// One 32B contiguous load per lane.
__device__ __forceinline__ v16bf load_a_frag(const unsigned short* row0, int ld, int k0) {
  int lane = threadIdx.x & 31;
  int r = lane & 15, hi = lane >> 4;
  const unsigned short* q = row0 + (size_t)r * ld + k0 + hi * 16;
  return *(const v16bf*)q;
}

// 16-bit B fragment 32x16: lane n<16 -> K k0..k0+15, lane n+16 -> k0+16..k0+31.
// One 32B contiguous load per lane from row-major B^T (N x K).
__device__ __forceinline__ v16bf load_b_frag(const unsigned short* col0, int ld, int k0) {
  int lane = threadIdx.x & 31;
  int n = lane & 15, hi = lane >> 4;
  const unsigned short* q = col0 + (size_t)n * ld + k0 + hi * 16;
  return *(const v16bf*)q;
}

__device__ __forceinline__ v8f wmma_bf16(v16bf a, v16bf b, v8f c) {
  return __builtin_amdgcn_wmma_f32_16x16x32_bf16(false, a, false, b, (short)0, c, false, false);
}

__device__ __forceinline__ v8f v8f_zero() {
  v8f z = {0.f, 0.f, 0.f, 0.f, 0.f, 0.f, 0.f, 0.f};
  return z;
}

// ---------- prep kernels ----------

__global__ void k_transpose(const float* __restrict__ fm, float* __restrict__ x) {
  size_t i = (size_t)blockIdx.x * blockDim.x + threadIdx.x;
  if (i >= (size_t)B_ * C_ * N_) return;
  int pix = (int)(i % N_);
  int t = (int)(i / N_);
  int c = t % C_;
  int b = t / C_;
  x[((size_t)(b * N_ + pix)) * C_ + c] = fm[i];
}

__global__ __launch_bounds__(256) void k_rownorm(const float* __restrict__ x,
                                                 unsigned short* __restrict__ xb,
                                                 unsigned short* __restrict__ nf) {
  __shared__ float red[256];
  int node = blockIdx.x;
  int tid = threadIdx.x;
  float v = x[(size_t)node * C_ + tid];
  red[tid] = v * v;
  __syncthreads();
  for (int s = 128; s > 0; s >>= 1) {
    if (tid < s) red[tid] += red[tid + s];
    __syncthreads();
  }
  float scale = 1.0f / fmaxf(sqrtf(red[0]), 1e-12f);
  xb[(size_t)node * C_ + tid] = f2bf(v);
  nf[(size_t)node * C_ + tid] = f2bf(v * scale);
}

__global__ void k_wtrans(const float* __restrict__ W, unsigned short* __restrict__ Wt,
                         int K, int Nf) {
  int i = blockIdx.x * blockDim.x + threadIdx.x;
  if (i >= K * Nf) return;
  int k = i / Nf, n = i % Nf;
  Wt[(size_t)n * K + k] = f2bf(W[i]);
}

__global__ void k_deginit(int* __restrict__ degi) {
  int i = blockIdx.x * blockDim.x + threadIdx.x;
  if (i < NTOT) degi[i] = 1;   // self loop
}

__global__ void k_dinv(const int* __restrict__ degi, float* __restrict__ dinv) {
  int i = blockIdx.x * blockDim.x + threadIdx.x;
  if (i < NTOT) dinv[i] = rsqrtf((float)degi[i]);
}

// ---------- similarity + fused top-k ----------

__device__ __forceinline__ void load_btile(v16bf (&b)[8], const unsigned short* bb) {
#pragma unroll
  for (int ks = 0; ks < 8; ++ks) b[ks] = load_b_frag(bb, C_, ks * 32);
}

// transpose the 16x16 wmma result through this wave's LDS tile, then lanes 0..15
// each maintain a register top-5 for one row
__device__ __forceinline__ void topk_scan(const v8f& acc, int ct,
                                          float (&tv)[5], int (&ti)[5],
                                          float (*tile)[17]) {
  int lane = threadIdx.x & 31;
  int n = lane & 15, hi = lane >> 4;
#pragma unroll
  for (int r = 0; r < 8; ++r) tile[r + 8 * hi][n] = acc[r];   // M = r + 8*hi
  __builtin_amdgcn_wave_barrier();
  if (lane < 16) {
    int colbase = ct * 16;
#pragma unroll
    for (int j = 0; j < 16; ++j) {
      float v = tile[lane][j];
      int id = colbase + j;
      if (v > tv[4]) {
        tv[4] = v; ti[4] = id;
#pragma unroll
        for (int q = 4; q > 0; --q) {
          if (tv[q] > tv[q - 1]) {
            float tf = tv[q]; tv[q] = tv[q - 1]; tv[q - 1] = tf;
            int tx = ti[q]; ti[q] = ti[q - 1]; ti[q - 1] = tx;
          }
        }
      }
    }
  }
  __builtin_amdgcn_wave_barrier();
}

__global__ __launch_bounds__(256) void k_sim_topk(const unsigned short* __restrict__ nf,
                                                  int* __restrict__ tgt,
                                                  int* __restrict__ degi) {
  __shared__ float tile[8][16][17];
  int wv = threadIdx.x >> 5;
  int lane = threadIdx.x & 31;
  int rt = blockIdx.x * 8 + wv;          // 0..1151 row tiles
  int b = rt / TPB;
  int rowbase = (rt % TPB) * 16;         // local row base inside batch
  int bN = b * N_;
  const unsigned short* nfb = nf + (size_t)bN * C_;
  float (*mytile)[17] = tile[wv];

  // keep all 8 A-fragments (16x256 row block) in registers
  v16bf a[8];
#pragma unroll
  for (int ks = 0; ks < 8; ++ks)
    a[ks] = load_a_frag(nfb + (size_t)rowbase * C_, C_, ks * 32);

  float tv[5]; int ti[5];
#pragma unroll
  for (int q = 0; q < 5; ++q) { tv[q] = -3.4e38f; ti[q] = -1; }

  // double-buffered B tiles; loads for tile t+1 overlap the top-k scan of tile t
  v16bf b0[8], b1[8];
  load_btile(b0, nfb);
#pragma unroll 1
  for (int ct = 0; ct < TPB; ct += 2) {
    // prefetch 2 tiles ahead (8KB, one cacheline per lane stride)
    if (ct + 2 < TPB) {
      const char* pf = (const char*)(nfb + (size_t)(ct + 2) * 16 * C_) + lane * 256;
      __builtin_prefetch(pf, 0, 1);
    }
    v8f acc = v8f_zero();
#pragma unroll
    for (int ks = 0; ks < 8; ++ks) acc = wmma_bf16(a[ks], b0[ks], acc);
    load_btile(b1, nfb + (size_t)(ct + 1) * 16 * C_);   // TPB even -> always valid
    topk_scan(acc, ct, tv, ti, mytile);

    acc = v8f_zero();
#pragma unroll
    for (int ks = 0; ks < 8; ++ks) acc = wmma_bf16(a[ks], b1[ks], acc);
    if (ct + 2 < TPB) load_btile(b0, nfb + (size_t)(ct + 2) * 16 * C_);
    topk_scan(acc, ct + 1, tv, ti, mytile);
  }

  if (lane < 16) {
    int m = rowbase + lane;          // local self index
    int g = bN + m;                  // global node id
    int selfpos = 0;                 // default: drop the max (mimics idx[:,1:])
#pragma unroll
    for (int q = 1; q < 5; ++q) if (ti[q] == m) selfpos = q;
    int o = 0;
#pragma unroll
    for (int q = 0; q < 5; ++q) {
      if (q != selfpos) {
        int t = bN + ti[q];
        tgt[(size_t)g * 4 + o] = t;
        atomicAdd(&degi[t], 1);
        ++o;
      }
    }
  }
}

// ---------- bf16 WMMA GEMM: C[M,Nf] = A[M,K] * Bt[Nf,K]^T ----------
// double-buffered over 64-wide k-steps so loads of step k+1 overlap WMMAs of step k

template <int CT>
__global__ __launch_bounds__(256) void k_gemm(const unsigned short* __restrict__ A,
                                              const unsigned short* __restrict__ Bt,
                                              float* __restrict__ Cout, int K, int Nf) {
  int wv = threadIdx.x >> 5;
  int lane = threadIdx.x & 31;
  int m0 = blockIdx.x * 16;
  const unsigned short* arow = A + (size_t)m0 * K;
  int n0 = wv * CT * 16;
  v8f acc[CT];
#pragma unroll
  for (int ct = 0; ct < CT; ++ct) acc[ct] = v8f_zero();

  v16bf af0 = load_a_frag(arow, K, 0);
  v16bf bf0[CT];
#pragma unroll
  for (int ct = 0; ct < CT; ++ct) bf0[ct] = load_b_frag(Bt + (size_t)(n0 + ct * 16) * K, K, 0);

#pragma unroll 1
  for (int ks = 0; ks < K; ks += 64) {          // K is a multiple of 64
    v16bf af1 = load_a_frag(arow, K, ks + 32);
    v16bf bf1[CT];
#pragma unroll
    for (int ct = 0; ct < CT; ++ct)
      bf1[ct] = load_b_frag(Bt + (size_t)(n0 + ct * 16) * K, K, ks + 32);
#pragma unroll
    for (int ct = 0; ct < CT; ++ct) acc[ct] = wmma_bf16(af0, bf0[ct], acc[ct]);

    if (ks + 64 < K) {
      af0 = load_a_frag(arow, K, ks + 64);
#pragma unroll
      for (int ct = 0; ct < CT; ++ct)
        bf0[ct] = load_b_frag(Bt + (size_t)(n0 + ct * 16) * K, K, ks + 64);
    }
#pragma unroll
    for (int ct = 0; ct < CT; ++ct) acc[ct] = wmma_bf16(af1, bf1[ct], acc[ct]);
  }

  int n = lane & 15, hi = lane >> 4;
#pragma unroll
  for (int ct = 0; ct < CT; ++ct)
#pragma unroll
    for (int r = 0; r < 8; ++r)
      Cout[(size_t)(m0 + r + 8 * hi) * Nf + n0 + ct * 16 + n] = acc[ct][r];
}

// ---------- GCN aggregation ----------

__global__ void k_initacc(const float* __restrict__ h, const float* __restrict__ dinv,
                          float* __restrict__ acc, int F) {
  size_t i = (size_t)blockIdx.x * blockDim.x + threadIdx.x;
  if (i >= (size_t)NTOT * F) return;
  int t = (int)(i / F);
  acc[i] = h[i] * dinv[t];            // self-loop message
}

__global__ __launch_bounds__(256) void k_scatter(const float* __restrict__ h,
                                                 const float* __restrict__ dinv,
                                                 const int* __restrict__ tgt,
                                                 float* __restrict__ acc, int F) {
  int s = blockIdx.x;
  float ds = dinv[s];
  int t0 = tgt[s * 4 + 0], t1 = tgt[s * 4 + 1];
  int t2 = tgt[s * 4 + 2], t3 = tgt[s * 4 + 3];
  for (int f = threadIdx.x; f < F; f += blockDim.x) {
    float v = h[(size_t)s * F + f] * ds;
    unsafeAtomicAdd(&acc[(size_t)t0 * F + f], v);
    unsafeAtomicAdd(&acc[(size_t)t1 * F + f], v);
    unsafeAtomicAdd(&acc[(size_t)t2 * F + f], v);
    unsafeAtomicAdd(&acc[(size_t)t3 * F + f], v);
  }
}

__global__ void k_finalize1(const float* __restrict__ acc, const float* __restrict__ dinv,
                            const float* __restrict__ bias, unsigned short* __restrict__ x1b) {
  size_t i = (size_t)blockIdx.x * blockDim.x + threadIdx.x;
  if (i >= (size_t)NTOT * F1_) return;
  int t = (int)(i / F1_), f = (int)(i % F1_);
  float v = fmaxf(dinv[t] * acc[i] + bias[f], 0.f);
  x1b[i] = f2bf(v);
}

__global__ void k_finalize2(const float* __restrict__ acc, const float* __restrict__ dinv,
                            const float* __restrict__ bias, float* __restrict__ out) {
  size_t i = (size_t)blockIdx.x * blockDim.x + threadIdx.x;
  if (i >= (size_t)NTOT * F2_) return;
  int t = (int)(i / F2_), f = (int)(i % F2_);
  float v = fmaxf(dinv[t] * acc[i] + bias[f], 0.f);
  int b = t / N_, pix = t % N_;
  out[((size_t)(b * F2_ + f)) * N_ + pix] = v;   // (B, F2, H, W)
}

// ---------- workspace layout ----------
constexpr size_t SZ_XF32 = (size_t)NTOT * C_ * 4;
constexpr size_t SZ_XBF  = (size_t)NTOT * C_ * 2;
constexpr size_t OFF_X    = 0;                                    // f32 x; reused as h2 later
constexpr size_t OFF_XB   = OFF_X + SZ_XF32;
constexpr size_t OFF_NF   = OFF_XB + SZ_XBF;
constexpr size_t OFF_W1T  = OFF_NF + SZ_XBF;
constexpr size_t OFF_W2T  = OFF_W1T + (size_t)C_ * F1_ * 2;
constexpr size_t OFF_TGT  = OFF_W2T + (size_t)F1_ * F2_ * 2;
constexpr size_t OFF_DEGI = OFF_TGT + (size_t)NTOT * 4 * 4;
constexpr size_t OFF_DINV = OFF_DEGI + (size_t)NTOT * 4;
constexpr size_t OFF_H1   = OFF_DINV + (size_t)NTOT * 4;
constexpr size_t OFF_ACC1 = OFF_H1 + (size_t)NTOT * F1_ * 4;      // acc2 reuses this
constexpr size_t OFF_X1B  = OFF_ACC1 + (size_t)NTOT * F1_ * 4;

extern "C" void kernel_launch(void* const* d_in, const int* in_sizes, int n_in,
                              void* d_out, int out_size, void* d_ws, size_t ws_size,
                              hipStream_t stream) {
  const float* fm = (const float*)d_in[0];
  const float* W1 = (const float*)d_in[1];
  const float* b1 = (const float*)d_in[2];
  const float* W2 = (const float*)d_in[3];
  const float* b2 = (const float*)d_in[4];
  float* out = (float*)d_out;
  char* ws = (char*)d_ws;

  float*          x    = (float*)(ws + OFF_X);
  unsigned short* xb   = (unsigned short*)(ws + OFF_XB);
  unsigned short* nf   = (unsigned short*)(ws + OFF_NF);
  unsigned short* w1t  = (unsigned short*)(ws + OFF_W1T);
  unsigned short* w2t  = (unsigned short*)(ws + OFF_W2T);
  int*            tgt  = (int*)(ws + OFF_TGT);
  int*            degi = (int*)(ws + OFF_DEGI);
  float*          dinv = (float*)(ws + OFF_DINV);
  float*          h1   = (float*)(ws + OFF_H1);
  float*          acc1 = (float*)(ws + OFF_ACC1);
  unsigned short* x1b  = (unsigned short*)(ws + OFF_X1B);
  float*          h2   = (float*)(ws + OFF_X);     // reuse (x dead after rownorm)
  float*          acc2 = (float*)(ws + OFF_ACC1);  // reuse (h1/acc1 layer-1 done)

  // prep
  k_transpose<<<(int)(((size_t)B_ * C_ * N_ + 255) / 256), 256, 0, stream>>>(fm, x);
  k_rownorm<<<NTOT, 256, 0, stream>>>(x, xb, nf);
  k_wtrans<<<(C_ * F1_ + 255) / 256, 256, 0, stream>>>(W1, w1t, C_, F1_);
  k_wtrans<<<(F1_ * F2_ + 255) / 256, 256, 0, stream>>>(W2, w2t, F1_, F2_);
  k_deginit<<<(NTOT + 255) / 256, 256, 0, stream>>>(degi);

  // similarity + top-k graph (87 GFLOP of bf16 WMMA, L2 resident)
  k_sim_topk<<<(B_ * TPB) / 8, 256, 0, stream>>>(nf, tgt, degi);
  k_dinv<<<(NTOT + 255) / 256, 256, 0, stream>>>(degi, dinv);

  // layer 1: h1 = x @ W1 ; aggregate ; relu -> x1b (bf16)
  k_gemm<4><<<NTOT / 16, 256, 0, stream>>>(xb, w1t, h1, C_, F1_);
  k_initacc<<<(int)(((size_t)NTOT * F1_ + 255) / 256), 256, 0, stream>>>(h1, dinv, acc1, F1_);
  k_scatter<<<NTOT, 256, 0, stream>>>(h1, dinv, tgt, acc1, F1_);
  k_finalize1<<<(int)(((size_t)NTOT * F1_ + 255) / 256), 256, 0, stream>>>(acc1, dinv, b1, x1b);

  // layer 2: h2 = x1 @ W2 ; aggregate ; relu -> out (B, F2, H, W)
  k_gemm<2><<<NTOT / 16, 256, 0, stream>>>(x1b, w2t, h2, F1_, F2_);
  k_initacc<<<(int)(((size_t)NTOT * F2_ + 255) / 256), 256, 0, stream>>>(h2, dinv, acc2, F2_);
  k_scatter<<<NTOT, 256, 0, stream>>>(h2, dinv, tgt, acc2, F2_);
  k_finalize2<<<(int)(((size_t)NTOT * F2_ + 255) / 256), 256, 0, stream>>>(acc2, dinv, b2, out);

  (void)in_sizes; (void)n_in; (void)out_size; (void)ws_size;
}
